// HybridTextClassifier_65481071406856
// MI455X (gfx1250) — matrix operand order repaired and checked by reading, not compiled, not run on hardware
//
#include <hip/hip_runtime.h>
#include <hip/hip_bf16.h>
#include <math.h>
#include <stdint.h>

// Model dims (fixed by reference)
#define BB   16
#define SS   256
#define EE   256
#define HH   8
#define LL   2
#define FF   1024
#define NCC  4
#define DKK  32
#define TT   (BB*SS)      // 4096 token rows
#define GNQ  1024         // 4 gates * NQ(256)

typedef __attribute__((ext_vector_type(16))) _Float16 v16h;
typedef __attribute__((ext_vector_type(8)))  float    v8f;

__device__ __forceinline__ v8f wmma_f16f32(v16h a, v16h b, v8f c) {
  // D = A(16x32 f16) * B(32x16 f16) + C(16x16 f32)
  return __builtin_amdgcn_wmma_f32_16x16x32_f16(false, a, false, b, (short)0, c, false, false);
}
__device__ __forceinline__ float sigmoidf_(float x) { return 1.0f / (1.0f + __expf(-x)); }

// ---------------------------------------------------------------------------
// Elementwise / setup kernels
// ---------------------------------------------------------------------------
__global__ void k_zero_state(float* c, _Float16* h16, int n) {
  int i = blockIdx.x * blockDim.x + threadIdx.x;
  if (i < n) { c[i] = 0.0f; h16[i] = (_Float16)0.0f; }
}

__global__ void k_embed(const int* x, const float* emb, _Float16* emb16) {
  int i = blockIdx.x * blockDim.x + threadIdx.x;       // over TT*EE
  if (i >= TT * EE) return;
  int t = i / EE, e = i - t * EE;
  emb16[i] = (_Float16)emb[(size_t)x[t] * EE + e];
}

// lstm_W (4, 512, 256) -> Wx[256][1024], Wh[256][1024]  (column block g*256+q)
__global__ void k_pack_lstm(const float* W, _Float16* wx, _Float16* wh) {
  int i = blockIdx.x * blockDim.x + threadIdx.x;       // over 4*512*256
  if (i >= 4 * 512 * 256) return;
  int g = i / (512 * 256);
  int r = (i / 256) % 512;
  int q = i % 256;
  _Float16 v = (_Float16)W[i];
  if (r < 256) wx[(size_t)r * GNQ + g * 256 + q] = v;
  else         wh[(size_t)(r - 256) * GNQ + g * 256 + q] = v;
}

__global__ void k_f32_to_f16(const float* a, _Float16* b, int n) {
  int i = blockIdx.x * blockDim.x + threadIdx.x;
  if (i < n) b[i] = (_Float16)a[i];
}

// ---------------------------------------------------------------------------
// Generic WMMA GEMM: C[M,N] = A[M,K] @ B[K,N], f16 in / f32 out, row-major.
// One wave per 16x16 C tile; per 32-wide K step the wave stages its 32x16 B
// tile into a private LDS slice with GLOBAL_LOAD_ASYNC_TO_LDS_B128 (ASYNCcnt),
// then reads B fragments from LDS while A comes in as two b128 global loads.
// ---------------------------------------------------------------------------
__global__ void k_gemm_nn(const _Float16* __restrict__ A, const _Float16* __restrict__ Bm,
                          float* __restrict__ C, int M, int N, int K,
                          int lda, int ldb, int ldc) {
  __shared__ __align__(16) _Float16 Bs[4][32 * 16];   // 1 KB per wave, 4 waves/block
  int wave = (blockIdx.x * blockDim.x + threadIdx.x) >> 5;
  int lane = threadIdx.x & 31;
  int wib  = (threadIdx.x >> 5) & 3;
  int tilesN = N >> 4;
  int tiles  = (M >> 4) * tilesN;
  if (wave >= tiles) return;
  int tm = wave / tilesN, tn = wave - tm * tilesN;

  int arow = lane & 15;
  int ak   = (lane >> 4) << 4;   // 0 or 16
  int bcol = lane & 15;
  int bk   = (lane >> 4) << 4;   // 0 or 16

  // async-copy lane mapping: each lane moves 16 B (8 halves)
  int crow  = lane >> 1;           // rows 0..15 (first op) / +16 (second op)
  int chalf = (lane & 1) * 8;      // which half-row
  _Float16* bs = &Bs[wib][0];
  uint32_t lds0 = (uint32_t)(uintptr_t)(bs + crow * 16 + chalf);
  uint32_t lds1 = (uint32_t)(uintptr_t)(bs + (crow + 16) * 16 + chalf);

  v8f acc = {};
  for (int k0 = 0; k0 < K; k0 += 32) {
    uint64_t g0 = (uint64_t)(uintptr_t)(Bm + (size_t)(k0 + crow) * ldb + tn * 16 + chalf);
    uint64_t g1 = (uint64_t)(uintptr_t)(Bm + (size_t)(k0 + 16 + crow) * ldb + tn * 16 + chalf);
    // drain pending LDS reads of the previous iteration, then kick the DMA
    asm volatile("s_wait_dscnt 0\n\t"
                 "global_load_async_to_lds_b128 %0, %1, off"
                 :: "v"(lds0), "v"(g0) : "memory");
    asm volatile("global_load_async_to_lds_b128 %0, %1, off"
                 :: "v"(lds1), "v"(g1) : "memory");

    const _Float16* ap = A + (size_t)(tm * 16 + arow) * lda + k0 + ak;
    if (k0 + 32 < K) __builtin_prefetch(ap + 32, 0, 0);   // global_prefetch_b8
    v16h a = {}, b = {};
#pragma unroll
    for (int i = 0; i < 16; ++i) a[i] = ap[i];

    asm volatile("s_wait_asynccnt 0" ::: "memory");       // B tile landed in LDS
#pragma unroll
    for (int i = 0; i < 16; ++i) b[i] = bs[(bk + i) * 16 + bcol];

    acc = wmma_f16f32(a, b, acc);
  }
  int cn = lane & 15;
  int mb = (lane >> 4) << 3;
  float* cp = C + (size_t)(tm * 16 + mb) * ldc + tn * 16 + cn;
#pragma unroll
  for (int r = 0; r < 8; ++r) cp[(size_t)r * ldc] = acc[r];
}

// ---------------------------------------------------------------------------
// Attention scores: per (b,h): S = Q(256x32) @ K^T  -> f16 scores (ldc = SS)
// B operand given as N x K rows (k-vectors contiguous) => NT form, one WMMA.
// ---------------------------------------------------------------------------
__global__ void k_attn_scores(const _Float16* __restrict__ q, const _Float16* __restrict__ kx,
                              _Float16* __restrict__ sc) {
  int bi = blockIdx.y;                 // b*HH + h
  int b = bi >> 3, hh = bi & 7;
  const _Float16* A  = q  + (size_t)b * SS * EE + hh * DKK;
  const _Float16* Bt = kx + (size_t)b * SS * EE + hh * DKK;
  _Float16* C = sc + (size_t)bi * SS * SS;

  int wave = (blockIdx.x * blockDim.x + threadIdx.x) >> 5;   // 0..255
  int lane = threadIdx.x & 31;
  int tm = wave >> 4, tn = wave & 15;

  int arow = lane & 15, ak = (lane >> 4) << 4;
  int bcol = lane & 15, bk = (lane >> 4) << 4;

  v16h a = {}, bb = {};
  const _Float16* ap = A + (size_t)(tm * 16 + arow) * EE + ak;
#pragma unroll
  for (int i = 0; i < 16; ++i) a[i] = ap[i];
  const _Float16* bp = Bt + (size_t)(tn * 16 + bcol) * EE + bk;
#pragma unroll
  for (int i = 0; i < 16; ++i) bb[i] = bp[i];
  v8f acc = {};
  acc = wmma_f16f32(a, bb, acc);

  int cn = lane & 15, mb = (lane >> 4) << 3;
  _Float16* cp = C + (size_t)(tm * 16 + mb) * SS + tn * 16 + cn;
#pragma unroll
  for (int r = 0; r < 8; ++r) cp[(size_t)r * SS] = (_Float16)acc[r];
}

// out(b,h) = attn(256x256) @ V(256x32), written into (TT x EE) at column h*32
__global__ void k_attn_av(const _Float16* __restrict__ attn, const _Float16* __restrict__ v,
                          float* __restrict__ out) {
  int bi = blockIdx.y;
  int b = bi >> 3, hh = bi & 7;
  const _Float16* A  = attn + (size_t)bi * SS * SS;           // lda = SS
  const _Float16* Bm = v + (size_t)b * SS * EE + hh * DKK;    // ldb = EE
  float* C = out + (size_t)b * SS * EE + hh * DKK;            // ldc = EE

  int wave = (blockIdx.x * blockDim.x + threadIdx.x) >> 5;    // 0..31
  int lane = threadIdx.x & 31;
  int tm = wave >> 1, tn = wave & 1;

  int arow = lane & 15, ak = (lane >> 4) << 4;
  int bcol = lane & 15, bk = (lane >> 4) << 4;

  v8f acc = {};
  for (int k0 = 0; k0 < SS; k0 += 32) {
    const _Float16* ap = A + (size_t)(tm * 16 + arow) * SS + k0 + ak;
    const _Float16* bp = Bm + (size_t)(k0 + bk) * EE + tn * 16 + bcol;
    v16h a = {}, bb = {};
#pragma unroll
    for (int i = 0; i < 16; ++i) a[i] = ap[i];
#pragma unroll
    for (int i = 0; i < 16; ++i) bb[i] = bp[(size_t)i * EE];
    acc = wmma_f16f32(a, bb, acc);
  }
  int cn = lane & 15, mb = (lane >> 4) << 3;
  float* cp = C + (size_t)(tm * 16 + mb) * EE + tn * 16 + cn;
#pragma unroll
  for (int r = 0; r < 8; ++r) cp[(size_t)r * EE] = acc[r];
}

// ---------------------------------------------------------------------------
// QLSTM per-timestep pointwise: z = zx[t] + zh + b; q = cumprod(cos(z+theta));
// gates -> c,h update. Single block of 256 threads (wave32 x 8).
// ---------------------------------------------------------------------------
__global__ void k_lstm_point(const float* __restrict__ zx, const float* __restrict__ zh,
                             const float* __restrict__ bgate, const float* __restrict__ theta,
                             float* __restrict__ qv, float* __restrict__ c,
                             float* __restrict__ hs, _Float16* __restrict__ h16, int t) {
  int tid = threadIdx.x;
  if (tid < 64) {                     // one thread per (gate g, batch b) row
    int g = tid >> 4, b = tid & 15;
    const float* zxr = zx + (size_t)(b * SS + t) * GNQ + g * 256;
    const float* zhr = zh + (size_t)b * GNQ + g * 256;
    const float* br  = bgate + g * 256;
    const float* th  = theta + g * 256;
    float* qr = qv + (size_t)(g * 16 + b) * 256;
    float p = 1.0f;
    for (int qi = 0; qi < 256; ++qi) {
      p *= cosf(zxr[qi] + zhr[qi] + br[qi] + th[qi]);
      qr[qi] = p;                     // cumulative product (MeasureAll Z expectations)
    }
  }
  __threadfence();
  __syncthreads();
#pragma unroll 4
  for (int it = 0; it < 16; ++it) {
    int idx = tid + it * 256;         // 0..4095 -> (b,q)
    int b = idx >> 8, q = idx & 255;
    float f  = sigmoidf_(qv[(size_t)(0 * 16 + b) * 256 + q]);
    float ii = sigmoidf_(qv[(size_t)(1 * 16 + b) * 256 + q]);
    float g  = tanhf   (qv[(size_t)(2 * 16 + b) * 256 + q]);
    float o  = sigmoidf_(qv[(size_t)(3 * 16 + b) * 256 + q]);
    float cn = f * c[idx] + ii * g;
    c[idx] = cn;
    float hn = o * tanhf(cn);
    hs[(size_t)(b * SS + t) * EE + q] = hn;
    h16[idx] = (_Float16)hn;
  }
}

// sinusoidal PE add (in place on hs)
__global__ void k_add_pe(float* h) {
  int i = blockIdx.x * blockDim.x + threadIdx.x;   // TT*EE
  if (i >= TT * EE) return;
  int row = i / EE, e = i - row * EE;
  int s = row % SS;
  int pair = e >> 1;
  float div = __expf((float)(2 * pair) * (-logf(10000.0f) / (float)EE));
  float ang = (float)s * div;
  h[i] += (e & 1) ? cosf(ang) : sinf(ang);
}

// qproj: out16[row] = cumprod(cos(h[row] + theta)) as f16
__global__ void k_qproj(const float* __restrict__ h, const float* __restrict__ theta,
                        _Float16* __restrict__ out16) {
  int r = blockIdx.x * blockDim.x + threadIdx.x;
  if (r >= TT) return;
  const float* hr = h + (size_t)r * EE;
  _Float16* orow = out16 + (size_t)r * EE;
  float p = 1.0f;
  for (int e = 0; e < EE; ++e) { p *= cosf(hr[e] + theta[e]); orow[e] = (_Float16)p; }
}

// softmax over 256 cols, in-place on f16 scores; one wave per row
__global__ void k_softmax_h(_Float16* sc, float scale, int rows) {
  int wave = (blockIdx.x * blockDim.x + threadIdx.x) >> 5;
  int lane = threadIdx.x & 31;
  if (wave >= rows) return;
  _Float16* row = sc + (size_t)wave * SS;
  float v[8];
  float m = -1e30f;
#pragma unroll
  for (int i = 0; i < 8; ++i) { v[i] = (float)row[lane + i * 32] * scale; m = fmaxf(m, v[i]); }
#pragma unroll
  for (int off = 16; off > 0; off >>= 1) m = fmaxf(m, __shfl_xor(m, off, 32));
  float s = 0.0f;
#pragma unroll
  for (int i = 0; i < 8; ++i) { v[i] = __expf(v[i] - m); s += v[i]; }
#pragma unroll
  for (int off = 16; off > 0; off >>= 1) s += __shfl_xor(s, off, 32);
  float inv = 1.0f / s;
#pragma unroll
  for (int i = 0; i < 8; ++i) row[lane + i * 32] = (_Float16)(v[i] * inv);
}

// h = LayerNorm(h + y + bias) * g + b, rowwise over EE; one wave per row (in place OK)
__global__ void k_bias_res_ln(const float* __restrict__ hin, const float* __restrict__ y,
                              const float* __restrict__ bias, const float* __restrict__ gam,
                              const float* __restrict__ bet, float* __restrict__ hout, int rows) {
  int wave = (blockIdx.x * blockDim.x + threadIdx.x) >> 5;
  int lane = threadIdx.x & 31;
  if (wave >= rows) return;
  const float* hr = hin + (size_t)wave * EE;
  const float* yr = y + (size_t)wave * EE;
  float v[8];
  float s = 0.0f;
#pragma unroll
  for (int i = 0; i < 8; ++i) { int e = lane + i * 32; v[i] = hr[e] + yr[e] + bias[e]; s += v[i]; }
#pragma unroll
  for (int off = 16; off > 0; off >>= 1) s += __shfl_xor(s, off, 32);
  float mean = s * (1.0f / EE);
  float var = 0.0f;
#pragma unroll
  for (int i = 0; i < 8; ++i) { float d = v[i] - mean; var += d * d; }
#pragma unroll
  for (int off = 16; off > 0; off >>= 1) var += __shfl_xor(var, off, 32);
  float inv = rsqrtf(var * (1.0f / EE) + 1e-5f);
  float* orow = hout + (size_t)wave * EE;
#pragma unroll
  for (int i = 0; i < 8; ++i) {
    int e = lane + i * 32;
    orow[e] = (v[i] - mean) * inv * gam[e] + bet[e];
  }
}

// FFN mid: relu16 = f16(max(y + bias, 0))
__global__ void k_bias_relu_h(const float* __restrict__ y, const float* __restrict__ bias,
                              _Float16* __restrict__ out16, int n, int cols) {
  int i = blockIdx.x * blockDim.x + threadIdx.x;
  if (i >= n) return;
  out16[i] = (_Float16)fmaxf(y[i] + bias[i % cols], 0.0f);
}

__global__ void k_pool(const float* __restrict__ h, float* __restrict__ pooled) {
  int i = blockIdx.x * blockDim.x + threadIdx.x;   // BB*EE
  if (i >= BB * EE) return;
  int b = i / EE, e = i - b * EE;
  float s = 0.0f;
  for (int t = 0; t < SS; ++t) s += h[(size_t)(b * SS + t) * EE + e];
  pooled[i] = s * (1.0f / SS);
}

__global__ void k_head(const float* __restrict__ pooled, const float* __restrict__ W,
                       const float* __restrict__ bias, float* __restrict__ out) {
  int i = blockIdx.x * blockDim.x + threadIdx.x;   // BB*NCC
  if (i >= BB * NCC) return;
  int b = i / NCC, c = i - b * NCC;
  float acc = bias[c];
  for (int e = 0; e < EE; ++e) acc += pooled[b * EE + e] * W[e * NCC + c];
  out[i] = acc;
}

// ---------------------------------------------------------------------------
// Host orchestration
// ---------------------------------------------------------------------------
extern "C" void kernel_launch(void* const* d_in, const int* in_sizes, int n_in,
                              void* d_out, int out_size, void* d_ws, size_t ws_size,
                              hipStream_t stream) {
  (void)in_sizes; (void)n_in; (void)out_size; (void)ws_size;

  const int*   x         = (const int*)  d_in[0];
  const float* token_emb = (const float*)d_in[1];
  const float* lstm_W    = (const float*)d_in[2];
  const float* lstm_b    = (const float*)d_in[3];
  const float* lstm_th   = (const float*)d_in[4];
  const float* ln1_g     = (const float*)d_in[5];
  const float* ln1_b     = (const float*)d_in[6];
  const float* ln2_g     = (const float*)d_in[7];
  const float* ln2_b     = (const float*)d_in[8];
  const float* qkv_theta = (const float*)d_in[9];
  const float* combine_W = (const float*)d_in[10];
  const float* combine_b = (const float*)d_in[11];
  const float* ffn_theta = (const float*)d_in[12];
  const float* lin1_W    = (const float*)d_in[13];
  const float* lin1_b    = (const float*)d_in[14];
  const float* lin2_W    = (const float*)d_in[15];
  const float* lin2_b    = (const float*)d_in[16];
  const float* cls_W     = (const float*)d_in[17];
  const float* cls_b     = (const float*)d_in[18];
  float* out = (float*)d_out;

  char* ws = (char*)d_ws;
  // ---- persistent ----
  float*    hs    = (float*)ws;                               // TT*EE f32 (4 MB)
  size_t oS = (size_t)TT * EE * 4;
  // ---- LSTM-phase scratch (overlaps attention scratch) ----
  _Float16* emb16 = (_Float16*)(ws + oS);                     // 2 MB
  float*    zx    = (float*)(ws + oS + (size_t)TT * EE * 2);  // 16 MB
  size_t oW = oS + (size_t)TT * EE * 2 + (size_t)TT * GNQ * 4;
  _Float16* wx16  = (_Float16*)(ws + oW);                     // 0.5 MB
  _Float16* wh16  = (_Float16*)(ws + oW + (size_t)256 * GNQ * 2);
  size_t oL = oW + (size_t)2 * 256 * GNQ * 2;
  _Float16* h16   = (_Float16*)(ws + oL);                     // 8 KB
  float*    cst   = (float*)(ws + oL + 16 * 256 * 2);         // 16 KB
  float*    zh    = (float*)(ws + oL + 16 * 256 * 2 + 16 * 256 * 4);   // 64 KB
  float*    qv    = (float*)(ws + oL + 16 * 256 * 2 + 16 * 256 * 4 + 16 * GNQ * 4); // 64 KB
  // ---- attention/FFN-phase scratch (reuses from oS) ----
  _Float16* q16   = (_Float16*)(ws + oS);
  _Float16* k16   = q16 + (size_t)TT * EE;
  _Float16* v16   = k16 + (size_t)TT * EE;
  _Float16* sc16  = v16 + (size_t)TT * EE;                    // BB*HH*SS*SS f16 (16.8 MB)
  float*    outb  = (float*)(sc16 + (size_t)BB * HH * SS * SS);   // TT*EE f32
  _Float16* out16 = (_Float16*)(outb + (size_t)TT * EE);
  _Float16* w16   = out16 + (size_t)TT * EE;                  // up to EE*FF halves
  float*    tmp1  = (float*)(w16 + (size_t)EE * FF);          // TT*FF f32 (16 MB)
  _Float16* relu16= (_Float16*)(tmp1 + (size_t)TT * FF);      // TT*FF f16 (8 MB)
  float*    tmp2  = (float*)(relu16 + (size_t)TT * FF);       // TT*EE f32 (4 MB)
  float*    pooled= (float*)(tmp2 + (size_t)TT * EE);         // BB*EE f32

  auto gemm = [&](const _Float16* A, const _Float16* Bm, float* C,
                  int M, int N, int K, int lda, int ldb, int ldc) {
    int tiles = (M / 16) * (N / 16);
    int blocks = (tiles + 3) / 4;                             // 128 threads = 4 waves
    k_gemm_nn<<<blocks, 128, 0, stream>>>(A, Bm, C, M, N, K, lda, ldb, ldc);
  };

  // ---- setup ----
  k_zero_state<<<(16 * 256 + 255) / 256, 256, 0, stream>>>(cst, h16, 16 * 256);
  k_embed<<<(TT * EE + 255) / 256, 256, 0, stream>>>(x, token_emb, emb16);
  k_pack_lstm<<<(4 * 512 * 256 + 255) / 256, 256, 0, stream>>>(lstm_W, wx16, wh16);

  // zx = emb @ Wx  (4096x256 @ 256x1024), hoisted out of the scan
  gemm(emb16, wx16, zx, TT, GNQ, EE, EE, GNQ, GNQ);

  // ---- QLSTM sequential scan (256 steps) ----
  for (int t = 0; t < SS; ++t) {
    gemm(h16, wh16, zh, 16, GNQ, EE, EE, GNQ, GNQ);           // zh = h @ Wh
    k_lstm_point<<<1, 256, 0, stream>>>(zx, zh, lstm_b, lstm_th, qv, cst, hs, h16, t);
  }

  k_add_pe<<<(TT * EE + 255) / 256, 256, 0, stream>>>(hs);

  // ---- transformer blocks ----
  const float scale = 1.0f / sqrtf((float)DKK);
  for (int l = 0; l < LL; ++l) {
    k_qproj<<<(TT + 255) / 256, 256, 0, stream>>>(hs, qkv_theta + (size_t)(l * 3 + 0) * EE, q16);
    k_qproj<<<(TT + 255) / 256, 256, 0, stream>>>(hs, qkv_theta + (size_t)(l * 3 + 1) * EE, k16);
    k_qproj<<<(TT + 255) / 256, 256, 0, stream>>>(hs, qkv_theta + (size_t)(l * 3 + 2) * EE, v16);

    k_attn_scores<<<dim3(64, BB * HH), 128, 0, stream>>>(q16, k16, sc16);
    k_softmax_h<<<(BB * HH * SS + 7) / 8, 256, 0, stream>>>(sc16, scale, BB * HH * SS);
    k_attn_av<<<dim3(8, BB * HH), 128, 0, stream>>>(sc16, v16, outb);

    k_f32_to_f16<<<(TT * EE + 255) / 256, 256, 0, stream>>>(outb, out16, TT * EE);
    k_f32_to_f16<<<(EE * EE + 255) / 256, 256, 0, stream>>>(combine_W + (size_t)l * EE * EE, w16, EE * EE);
    gemm(out16, w16, tmp2, TT, EE, EE, EE, EE, EE);
    k_bias_res_ln<<<(TT + 7) / 8, 256, 0, stream>>>(hs, tmp2, combine_b + (size_t)l * EE,
                                                    ln1_g + (size_t)l * EE, ln1_b + (size_t)l * EE,
                                                    hs, TT);

    // FFN with quantum projection front-end
    k_qproj<<<(TT + 255) / 256, 256, 0, stream>>>(hs, ffn_theta + (size_t)l * EE, q16);
    k_f32_to_f16<<<(EE * FF + 255) / 256, 256, 0, stream>>>(lin1_W + (size_t)l * EE * FF, w16, EE * FF);
    gemm(q16, w16, tmp1, TT, FF, EE, EE, FF, FF);
    k_bias_relu_h<<<(TT * FF + 255) / 256, 256, 0, stream>>>(tmp1, lin1_b + (size_t)l * FF,
                                                             relu16, TT * FF, FF);
    k_f32_to_f16<<<(FF * EE + 255) / 256, 256, 0, stream>>>(lin2_W + (size_t)l * FF * EE, w16, FF * EE);
    gemm(relu16, w16, tmp2, TT, EE, FF, FF, EE, EE);
    k_bias_res_ln<<<(TT + 7) / 8, 256, 0, stream>>>(hs, tmp2, lin2_b + (size_t)l * EE,
                                                    ln2_g + (size_t)l * EE, ln2_b + (size_t)l * EE,
                                                    hs, TT);
  }

  // ---- pool + classifier ----
  k_pool<<<(BB * EE + 255) / 256, 256, 0, stream>>>(hs, pooled);
  k_head<<<1, 64, 0, stream>>>(pooled, cls_W, cls_b, out);
}
